// Attention_82703890252416
// MI455X (gfx1250) — matrix operand order repaired and checked by reading, not compiled, not run on hardware
//
#include <hip/hip_runtime.h>
#include <hip/hip_bf16.h>

typedef __bf16 bf16_t;
typedef bf16_t v16bf __attribute__((ext_vector_type(16)));
typedef bf16_t v8bf  __attribute__((ext_vector_type(8)));
typedef float  v8f   __attribute__((ext_vector_type(8)));
typedef int    v4i   __attribute__((vector_size(16)));   // matches builtin's V4i

#define NB  4
#define NT  2048
#define ND  1152
#define NH  8
#define NHD 144
#define HDP 160           // NHD padded to multiple of 32 for K-dim of QK^T
#define NM  (NB*NT)       // 8192 tokens

// ---------- async copy to LDS (gfx1250) with sync fallback ----------

#if defined(__gfx1250__) && __has_builtin(__builtin_amdgcn_global_load_async_to_lds_b128)
#define USE_ASYNC 1
#else
#define USE_ASYNC 0
#endif

__device__ __forceinline__ void cp_b128(void* lds, const void* g) {
#if USE_ASYNC
  __builtin_amdgcn_global_load_async_to_lds_b128(
      (__attribute__((address_space(1))) v4i*)g,
      (__attribute__((address_space(3))) v4i*)lds, 0, 0);
#else
  *(v8bf*)lds = *(const v8bf*)g;
#endif
}

__device__ __forceinline__ void cp_wait() {
#if USE_ASYNC
#if __has_builtin(__builtin_amdgcn_s_wait_asynccnt)
  __builtin_amdgcn_s_wait_asynccnt(0);
#else
  asm volatile("s_wait_asynccnt 0x0" ::: "memory");
#endif
#endif
}

// ---------- helpers ----------

__device__ __forceinline__ bf16_t f2bf(float f) {
  union { float f; unsigned u; } v; v.f = f;
  unsigned r = v.u + 0x7FFFu + ((v.u >> 16) & 1u);   // round-to-nearest-even
  unsigned short s = (unsigned short)(r >> 16);
  bf16_t o; __builtin_memcpy(&o, &s, 2);
  return o;
}

// A-fragment (16x32 bf16, M x K): lane holds row (lane&15); elements 0..7 are
// k = half*8 .. half*8+7, elements 8..15 are k = 16+half*8 .. (ISA 7.12.2).
__device__ __forceinline__ v16bf load_fragA(const bf16_t* p, int half) {
  const bf16_t* q = p + half * 8;
  v8bf lo = *(const v8bf*)(q);
  v8bf hi = *(const v8bf*)(q + 16);
  return __builtin_shufflevector(lo, hi, 0,1,2,3,4,5,6,7,8,9,10,11,12,13,14,15);
}

// B-fragment (32x16 bf16, K x N): lane holds column (lane&15); elements 0..7 are
// k = half*8 .. half*8+7, elements 8..15 are k = 16+half*8 .. (same as A).
__device__ __forceinline__ v16bf load_fragB(const bf16_t* p, int half) {
  const bf16_t* q = p + half * 8;
  v8bf lo = *(const v8bf*)(q);
  v8bf hi = *(const v8bf*)(q + 16);
  return __builtin_shufflevector(lo, hi, 0,1,2,3,4,5,6,7,8,9,10,11,12,13,14,15);
}

__device__ __forceinline__ v8f wmma_bf16(v16bf a, v16bf b, v8f c) {
  return __builtin_amdgcn_wmma_f32_16x16x32_bf16(
      /*neg_a=*/false, a, /*neg_b=*/false, b,
      /*c_mod=*/(short)0, c, /*reuse_a=*/false, /*reuse_b=*/false);
}

// ---------- elementwise cast f32 -> bf16 (x4 per thread) ----------

__global__ void cast_kernel(const float* __restrict__ src, bf16_t* __restrict__ dst, int n) {
  int i = (blockIdx.x * 256 + threadIdx.x) * 4;
  if (i < n) {
    float4 f = *(const float4*)(src + i);
    dst[i + 0] = f2bf(f.x);
    dst[i + 1] = f2bf(f.y);
    dst[i + 2] = f2bf(f.z);
    dst[i + 3] = f2bf(f.w);
  }
}

// ---------- GEMM: C[M=8192][N=1152] = A_bf16[M][1152] @ W_bf16[N][1152]^T ----------
// 128x128 block tile, LDS-staged and double-buffered via async copies.
// 8 waves; each computes 32x64 (2 A-frag x 4 B-frag -> 8 WMMAs / k-step).
// MODE 0: write f32 row-major to Cf.  MODE 1: write bf16 V^T [B][H][HD][T].

template<int MODE>
__global__ void __launch_bounds__(256) gemm_kernel(const bf16_t* __restrict__ A,
                                                   const bf16_t* __restrict__ W,
                                                   float* __restrict__ Cf,
                                                   bf16_t* __restrict__ Cvt) {
  const int tid  = threadIdx.x;
  const int lane = tid & 31;
  const int wave = tid >> 5;
  const int l15  = lane & 15;
  const int half = lane >> 4;
  const int wr = wave & 3, wc = wave >> 2;
  const int m0 = blockIdx.y * 128;
  const int n0 = blockIdx.x * 128;

  __shared__ __align__(16) bf16_t ldsA[2][128][32];
  __shared__ __align__(16) bf16_t ldsB[2][128][32];

  auto issue = [&](int buf, int kk) {
    // A tile: 128 rows x 64B = 512 x 16B chunks; same for B. 4 chunks/thread.
    int c = tid;
    int r = c >> 2, o = (c & 3) * 8;
    cp_b128(&ldsA[buf][r][o], A + (size_t)(m0 + r) * ND + kk + o);
    cp_b128(&ldsB[buf][r][o], W + (size_t)(n0 + r) * ND + kk + o);
    c = tid + 256;
    r = c >> 2; o = (c & 3) * 8;
    cp_b128(&ldsA[buf][r][o], A + (size_t)(m0 + r) * ND + kk + o);
    cp_b128(&ldsB[buf][r][o], W + (size_t)(n0 + r) * ND + kk + o);
  };

  v8f acc[2][4];
  #pragma unroll
  for (int s = 0; s < 2; ++s)
    #pragma unroll
    for (int j = 0; j < 4; ++j) { v8f z = {0,0,0,0,0,0,0,0}; acc[s][j] = z; }

  issue(0, 0);
  cp_wait();
  __syncthreads();

  constexpr int KSTEPS = ND / 32;   // 36
  for (int ki = 0; ki < KSTEPS; ++ki) {
    const int buf = ki & 1;
    if (ki + 1 < KSTEPS) issue(buf ^ 1, (ki + 1) * 32);

    v16bf a0 = load_fragA(&ldsA[buf][wr * 32 + l15][0], half);
    v16bf a1 = load_fragA(&ldsA[buf][wr * 32 + 16 + l15][0], half);
    #pragma unroll
    for (int j = 0; j < 4; ++j) {
      v16bf bfrag = load_fragB(&ldsB[buf][wc * 64 + j * 16 + l15][0], half);
      acc[0][j] = wmma_bf16(a0, bfrag, acc[0][j]);
      acc[1][j] = wmma_bf16(a1, bfrag, acc[1][j]);
    }
    cp_wait();        // next tile resident before anyone reads it
    __syncthreads();  // and before anyone overwrites the tile just read
  }

  #pragma unroll
  for (int s = 0; s < 2; ++s) {
    #pragma unroll
    for (int j = 0; j < 4; ++j) {
      const int n = n0 + wc * 64 + j * 16 + l15;
      if (MODE == 0) {
        #pragma unroll
        for (int r = 0; r < 8; ++r) {
          const int m = m0 + wr * 32 + s * 16 + r + 8 * half;
          Cf[(size_t)m * ND + n] = acc[s][j][r];
        }
      } else {
        const int hh = n / NHD, hd = n % NHD;
        #pragma unroll
        for (int r = 0; r < 8; ++r) {
          const int m = m0 + wr * 32 + s * 16 + r + 8 * half;
          const int bb = m >> 11, tt = m & (NT - 1);
          Cvt[((size_t)(bb * NH + hh) * NHD + hd) * NT + tt] = f2bf(acc[s][j][r]);
        }
      }
    }
  }
}

// ---------- rotary: f32 Q/K [M][1152] -> bf16 [B][H][T][HDP] (padded) ----------

__global__ void rotary_kernel(const float* __restrict__ Qf, const float* __restrict__ Kf,
                              const float* __restrict__ cosp, const float* __restrict__ sinp,
                              bf16_t* __restrict__ qb, bf16_t* __restrict__ kb) {
  int idx = blockIdx.x * 256 + threadIdx.x;
  const int total = NB * NT * NH * 72;
  if (idx >= total) return;
  int j = idx % 72;
  int h = (idx / 72) % NH;
  int t = (idx / (72 * NH)) % NT;
  int b = idx / (72 * NH * NT);

  size_t in0 = (size_t)(b * NT + t) * ND + h * NHD;
  float c1 = cosp[t * NHD + j],      s1 = sinp[t * NHD + j];
  float c2 = cosp[t * NHD + j + 72], s2 = sinp[t * NHD + j + 72];
  float q1 = Qf[in0 + j], q2 = Qf[in0 + j + 72];
  float k1 = Kf[in0 + j], k2 = Kf[in0 + j + 72];

  size_t ob0 = (size_t)((b * NH + h) * NT + t) * HDP;
  qb[ob0 + j]      = f2bf(q1 * c1 - q2 * s1);
  qb[ob0 + j + 72] = f2bf(q2 * c2 + q1 * s2);
  kb[ob0 + j]      = f2bf(k1 * c1 - k2 * s1);
  kb[ob0 + j + 72] = f2bf(k2 * c2 + k1 * s2);
  if (j < 16) {                       // zero the 144..159 pad
    qb[ob0 + 144 + j] = f2bf(0.f);
    kb[ob0 + 144 + j] = f2bf(0.f);
  }
}

// ---------- flash attention ----------
// q,k [B][H][T][HDP], vT [B][H][HD][T] -> ob [B][T][D]
// 4 waves/block, each owns 16 query rows. K/V tiles for each 32-key step are
// shared by all waves: async-copied to LDS, double-buffered.

__global__ void __launch_bounds__(128) flash_kernel(const bf16_t* __restrict__ qb,
                                                    const bf16_t* __restrict__ kbp,
                                                    const bf16_t* __restrict__ vT,
                                                    bf16_t* __restrict__ ob) {
  const int tid  = threadIdx.x;
  const int lane = tid & 31;
  const int wave = tid >> 5;
  const int l15  = lane & 15;
  const int half = lane >> 4;
  const int h = blockIdx.y, b = blockIdx.z;
  const int bh = b * NH + h;
  const int t0 = blockIdx.x * 64 + wave * 16;
  const float scale = 0.0833333333f; // 144^-0.5 = 1/12

  __shared__ __align__(16) bf16_t ldsK[2][32][HDP];   // 20.0 KB
  __shared__ __align__(16) bf16_t ldsV[2][NHD][32];   // 18.0 KB
  __shared__ __align__(16) float  lds_S[4][16][32];   //  8.0 KB
  __shared__ __align__(16) bf16_t lds_P[4][16][32];   //  4.0 KB
  __shared__ float lds_c[4][16];

  auto issue_kv = [&](int buf, int kt) {
    // K tile: 32 rows x 320B = 640 x 16B chunks (5/thread)
    for (int c = tid; c < 640; c += 128) {
      int r = c / 20, o = (c % 20) * 8;
      cp_b128(&ldsK[buf][r][o], kbp + (size_t)(bh * NT + kt + r) * HDP + o);
    }
    // V tile: 144 rows x 64B = 576 x 16B chunks
    for (int c = tid; c < 576; c += 128) {
      int r = c >> 2, o = (c & 3) * 8;
      cp_b128(&ldsV[buf][r][o], vT + (size_t)(bh * NHD + r) * NT + kt + o);
    }
  };

  const bf16_t* qrow = qb + (size_t)(bh * NT + t0 + l15) * HDP;
  v16bf qf[5];
  #pragma unroll
  for (int ks = 0; ks < 5; ++ks) qf[ks] = load_fragA(qrow + ks * 32, half);

  v8f oacc[9];
  #pragma unroll
  for (int jt = 0; jt < 9; ++jt) { v8f z = {0,0,0,0,0,0,0,0}; oacc[jt] = z; }

  float m_r = -3.0e38f, l_r = 0.f;   // valid in lanes 0..15 (row owners)

  issue_kv(0, 0);
  cp_wait();
  __syncthreads();

  for (int ki = 0; ki < NT / 32; ++ki) {
    const int kt = ki * 32;
    const int buf = ki & 1;
    if (kt + 32 < NT) issue_kv(buf ^ 1, kt + 32);

    // S = q @ k^T for a 16x32 tile, accumulated over HDP=160 (5 k-steps)
    v8f s0 = {0,0,0,0,0,0,0,0}, s1 = {0,0,0,0,0,0,0,0};
    #pragma unroll
    for (int ks = 0; ks < 5; ++ks) {
      v16bf k0 = load_fragB(&ldsK[buf][l15][ks * 32], half);
      v16bf k1 = load_fragB(&ldsK[buf][16 + l15][ks * 32], half);
      s0 = wmma_bf16(qf[ks], k0, s0);
      s1 = wmma_bf16(qf[ks], k1, s1);
    }
    #pragma unroll
    for (int r = 0; r < 8; ++r) {
      lds_S[wave][r + 8 * half][l15]      = s0[r] * scale;
      lds_S[wave][r + 8 * half][16 + l15] = s1[r] * scale;
    }
    __syncthreads();

    // online softmax per row (lane r owns row r)
    if (lane < 16) {
      float mx = m_r;
      for (int c = 0; c < 32; ++c) mx = fmaxf(mx, lds_S[wave][lane][c]);
      float corr = __expf(m_r - mx);
      float sum = 0.f;
      for (int c = 0; c < 32; ++c) {
        float p = __expf(lds_S[wave][lane][c] - mx);
        sum += p;
        lds_P[wave][lane][c] = f2bf(p);
      }
      l_r = l_r * corr + sum;
      m_r = mx;
      lds_c[wave][lane] = corr;
    }
    __syncthreads();

    float cr[8];
    #pragma unroll
    for (int r = 0; r < 8; ++r) cr[r] = lds_c[wave][r + 8 * half];

    v16bf pf = load_fragA(&lds_P[wave][l15][0], half);   // P as A-fragment
    #pragma unroll
    for (int jt = 0; jt < 9; ++jt) {
      v16bf vb = load_fragB(&ldsV[buf][jt * 16 + l15][0], half);
      #pragma unroll
      for (int r = 0; r < 8; ++r) oacc[jt][r] *= cr[r];
      oacc[jt] = wmma_bf16(pf, vb, oacc[jt]);
    }
    cp_wait();
    __syncthreads();
  }

  if (lane < 16) lds_c[wave][lane] = 1.f / l_r;
  __syncthreads();
  float inv[8];
  #pragma unroll
  for (int r = 0; r < 8; ++r) inv[r] = lds_c[wave][r + 8 * half];

  #pragma unroll
  for (int jt = 0; jt < 9; ++jt) {
    #pragma unroll
    for (int r = 0; r < 8; ++r) {
      int t = t0 + r + 8 * half;
      int n = h * NHD + jt * 16 + l15;
      ob[(size_t)(b * NT + t) * ND + n] = f2bf(oacc[jt][r] * inv[r]);
    }
  }
}

// ---------- host launch ----------

extern "C" void kernel_launch(void* const* d_in, const int* in_sizes, int n_in,
                              void* d_out, int out_size, void* d_ws, size_t ws_size,
                              hipStream_t stream) {
  (void)in_sizes; (void)n_in; (void)out_size; (void)ws_size;
  const float* x    = (const float*)d_in[0];
  const float* cosp = (const float*)d_in[1];
  const float* sinp = (const float*)d_in[2];
  const float* Wq   = (const float*)d_in[3];
  const float* Wk   = (const float*)d_in[4];
  const float* Wv   = (const float*)d_in[5];
  const float* Wo   = (const float*)d_in[6];
  float* out = (float*)d_out;
  char* ws = (char*)d_ws;

  constexpr size_t SZ_XB = (size_t)NM * ND * 2;
  constexpr size_t SZ_WB = (size_t)ND * ND * 2;
  constexpr size_t SZ_QF = (size_t)NM * ND * 4;
  constexpr size_t SZ_VT = (size_t)NB * NH * NHD * NT * 2;
  constexpr size_t SZ_QB = (size_t)NB * NH * NT * HDP * 2;

  size_t o_xb = 0;
  size_t o_wq = o_xb + SZ_XB;
  size_t o_wk = o_wq + SZ_WB;
  size_t o_wv = o_wk + SZ_WB;
  size_t o_wo = o_wv + SZ_WB;
  size_t o_qf = o_wo + SZ_WB;
  size_t o_kf = o_qf + SZ_QF;
  size_t o_vt = o_kf + SZ_QF;
  size_t o_qb = o_vt + SZ_VT;
  size_t o_kb = o_qb + SZ_QB;   // end ~ 166 MB

  bf16_t* xb  = (bf16_t*)(ws + o_xb);
  bf16_t* wqb = (bf16_t*)(ws + o_wq);
  bf16_t* wkb = (bf16_t*)(ws + o_wk);
  bf16_t* wvb = (bf16_t*)(ws + o_wv);
  bf16_t* wob = (bf16_t*)(ws + o_wo);
  float*  Qf  = (float*)(ws + o_qf);
  float*  Kf  = (float*)(ws + o_kf);
  bf16_t* vT  = (bf16_t*)(ws + o_vt);
  bf16_t* qb  = (bf16_t*)(ws + o_qb);
  bf16_t* kb  = (bf16_t*)(ws + o_kb);
  bf16_t* obf = (bf16_t*)(ws + o_qf); // reuse Qf region after rotary consumed it

  const int nX = NM * ND, nW = ND * ND;
  cast_kernel<<<dim3((nX / 4 + 255) / 256), dim3(256), 0, stream>>>(x,  xb,  nX);
  cast_kernel<<<dim3((nW / 4 + 255) / 256), dim3(256), 0, stream>>>(Wq, wqb, nW);
  cast_kernel<<<dim3((nW / 4 + 255) / 256), dim3(256), 0, stream>>>(Wk, wkb, nW);
  cast_kernel<<<dim3((nW / 4 + 255) / 256), dim3(256), 0, stream>>>(Wv, wvb, nW);
  cast_kernel<<<dim3((nW / 4 + 255) / 256), dim3(256), 0, stream>>>(Wo, wob, nW);

  dim3 gg(ND / 128, NM / 128);   // (9, 64)
  gemm_kernel<0><<<gg, dim3(256), 0, stream>>>(xb, wqb, Qf, nullptr);
  gemm_kernel<0><<<gg, dim3(256), 0, stream>>>(xb, wkb, Kf, nullptr);
  gemm_kernel<1><<<gg, dim3(256), 0, stream>>>(xb, wvb, nullptr, vT);

  const int nR = NB * NT * NH * 72;
  rotary_kernel<<<dim3((nR + 255) / 256), dim3(256), 0, stream>>>(Qf, Kf, cosp, sinp, qb, kb);

  flash_kernel<<<dim3(NT / 64, NH, NB), dim3(128), 0, stream>>>(qb, kb, vT, obf);

  gemm_kernel<0><<<gg, dim3(256), 0, stream>>>(obf, wob, out, nullptr);
}
